// SAM_58265526337603
// MI455X (gfx1250) — matrix operand order repaired and checked by reading, not compiled, not run on hardware
//
#include <hip/hip_runtime.h>

// ---------------------------------------------------------------------------
// MI455X (gfx1250) implementation.
// Roofline: workload is HBM-bound (~GB-scale intermediates @ 23.3 TB/s),
// so all intermediates are stored f16 (halves traffic), conv1x1 GEMMs use
// v_wmma_f32_16x16x32_f16 (f32 accumulate), and the final conv+relu+proj is
// fused so the last 32-ch tensor never touches HBM. wave32 everywhere.
// ---------------------------------------------------------------------------

typedef __attribute__((ext_vector_type(16))) _Float16 v16h;
typedef __attribute__((ext_vector_type(2)))  _Float16 v2h;
typedef __attribute__((ext_vector_type(8)))  float    v8f;

#define HH   256
#define WW   256
#define NB   4
#define CIN  64
#define COUT 32
#define PB   (HH * WW)     // 65536 pixels per batch image
#define PT   (NB * PB)     // 262144 total pixels

union FragA { v16h v; uint4 q[2]; };

// ---------------------------------------------------------------------------
// conv1: out1[m][p] = f16( sum_ci w1[m][ci] * x[ci][p] + b1[m] )
// Block: 256 thr (8 waves), 128-pixel tile, K=64 in two WMMA K-steps.
// ---------------------------------------------------------------------------
__global__ void __launch_bounds__(256) sam_conv1(
    const float* __restrict__ x, const float* __restrict__ w1,
    const float* __restrict__ b1, _Float16* __restrict__ out)
{
  __shared__ _Float16 sA[COUT * 64];   // [m][k], pitch 64 halves (128B)
  __shared__ _Float16 sB[128 * 72];    // [p][k], pitch 72 halves (144B, 16B mult)
  const int t = threadIdx.x;

  for (int i = t; i < COUT * CIN; i += 256)
    sA[(i / CIN) * 64 + (i % CIN)] = (_Float16)w1[i];

  const int p0  = blockIdx.x * 128;
  const int b   = p0 / PB;
  const int pp0 = p0 % PB;
  const float* xb = x + (size_t)b * CIN * PB + pp0;
  {
    const int j2 = t & 63;   // pixel-pair index (128 px)
    const int r0 = t >> 6;   // 4 channel rows per pass
    for (int pass = 0; pass < 16; ++pass) {
      int ci = r0 + 4 * pass;
      float2 f = *reinterpret_cast<const float2*>(xb + (size_t)ci * PB + 2 * j2);
      sB[(2 * j2)     * 72 + ci] = (_Float16)f.x;
      sB[(2 * j2 + 1) * 72 + ci] = (_Float16)f.y;
    }
  }
  __syncthreads();

  const int lane = t & 31, wv = t >> 5;
  const int n = lane & 15, hi = lane >> 4;
  const int pw = wv * 16;
  FragA a0, a1, bb;
  v8f acc0 = {}, acc1 = {};
  for (int kk = 0; kk < CIN; kk += 32) {
    const _Float16* ar0 = &sA[n * 64 + kk + hi * 8];
    const _Float16* ar1 = &sA[(n + 16) * 64 + kk + hi * 8];
    a0.q[0] = *(const uint4*)(ar0);      a0.q[1] = *(const uint4*)(ar0 + 16);
    a1.q[0] = *(const uint4*)(ar1);      a1.q[1] = *(const uint4*)(ar1 + 16);
    const _Float16* br = &sB[(pw + n) * 72 + kk + hi * 8];
    bb.q[0] = *(const uint4*)(br);       bb.q[1] = *(const uint4*)(br + 16);
    acc0 = __builtin_amdgcn_wmma_f32_16x16x32_f16(false, a0.v, false, bb.v,
                                                  (short)0, acc0, false, false);
    acc1 = __builtin_amdgcn_wmma_f32_16x16x32_f16(false, a1.v, false, bb.v,
                                                  (short)0, acc1, false, false);
  }
  const int gp = p0 + pw + n;
#pragma unroll
  for (int r = 0; r < 8; ++r) {
    int m0 = r + hi * 8;
    out[(size_t)m0 * PT + gp]        = (_Float16)(acc0[r] + b1[m0]);
    out[(size_t)(m0 + 16) * PT + gp] = (_Float16)(acc1[r] + b1[m0 + 16]);
  }
}

// ---------------------------------------------------------------------------
// convk<FUSED>: GEMM W[32][256] x cat[256][P]; FUSED adds bias+relu, dots with
// w_out and writes the f32 mask (cross half-wave reduce via shfl_xor 16).
// Block: 128 thr (4 waves), 64-pixel tile, K=256 in 8 WMMA K-steps.
// ---------------------------------------------------------------------------
template <bool FUSED>
__global__ void __launch_bounds__(128) sam_convk(
    const _Float16* __restrict__ in, const float* __restrict__ w,
    const float* __restrict__ bias, const float* __restrict__ wout,
    _Float16* __restrict__ out16, float* __restrict__ outf)
{
  __shared__ _Float16 sA[COUT * 264];  // [m][k], pitch 264 halves (528B)
  __shared__ _Float16 sB[64 * 264];    // [p][k], pitch 264
  const int t = threadIdx.x;

  for (int i = t; i < COUT * 256; i += 128)
    sA[(i >> 8) * 264 + (i & 255)] = (_Float16)w[i];

  const int p0 = blockIdx.x * 64;
  {
    const int pr = t & 31;   // pixel-pair index (64 px)
    const int r0 = t >> 5;   // 4 channel rows per pass
    for (int pass = 0; pass < 64; ++pass) {
      int k = r0 + 4 * pass;
      v2h u = *reinterpret_cast<const v2h*>(in + (size_t)k * PT + p0 + 2 * pr);
      sB[(2 * pr)     * 264 + k] = u.x;
      sB[(2 * pr + 1) * 264 + k] = u.y;
    }
  }
  __syncthreads();

  const int lane = t & 31, wv = t >> 5;
  const int n = lane & 15, hi = lane >> 4;
  const int pw = wv * 16;
  FragA a0, a1, bb;
  v8f acc0 = {}, acc1 = {};
#pragma unroll
  for (int kk = 0; kk < 256; kk += 32) {
    const _Float16* ar0 = &sA[n * 264 + kk + hi * 8];
    const _Float16* ar1 = &sA[(n + 16) * 264 + kk + hi * 8];
    a0.q[0] = *(const uint4*)(ar0);      a0.q[1] = *(const uint4*)(ar0 + 16);
    a1.q[0] = *(const uint4*)(ar1);      a1.q[1] = *(const uint4*)(ar1 + 16);
    const _Float16* br = &sB[(pw + n) * 264 + kk + hi * 8];
    bb.q[0] = *(const uint4*)(br);       bb.q[1] = *(const uint4*)(br + 16);
    acc0 = __builtin_amdgcn_wmma_f32_16x16x32_f16(false, a0.v, false, bb.v,
                                                  (short)0, acc0, false, false);
    acc1 = __builtin_amdgcn_wmma_f32_16x16x32_f16(false, a1.v, false, bb.v,
                                                  (short)0, acc1, false, false);
  }
  const int gp = p0 + pw + n;
  if (FUSED) {
    float s = 0.f;
#pragma unroll
    for (int r = 0; r < 8; ++r) {
      int m0 = r + hi * 8;
      s += wout[m0]      * fmaxf(acc0[r] + bias[m0], 0.f);
      s += wout[m0 + 16] * fmaxf(acc1[r] + bias[m0 + 16], 0.f);
    }
    s += __shfl_xor(s, 16);
    if (hi == 0) outf[gp] = s;
  } else {
#pragma unroll
    for (int r = 0; r < 8; ++r) {
      int m0 = r + hi * 8;
      out16[(size_t)m0 * PT + gp]        = (_Float16)(acc0[r] + bias[m0]);
      out16[(size_t)(m0 + 16) * PT + gp] = (_Float16)(acc1[r] + bias[m0 + 16]);
    }
  }
}

// ---------------------------------------------------------------------------
// IRNN scans. cat channel layout: (dir*32 + c) with dir order
// up=0, right=1, down=2, left=3, zuoxia=4, youxia=5, zuoshang=6, youshang=7.
// ---------------------------------------------------------------------------

// Forward H scan: down (wd), zuoxia (shift +1, wd), youxia (shift -1, wd).
__global__ void __launch_bounds__(256) sam_scan_h_fwd(
    const _Float16* __restrict__ in, _Float16* __restrict__ cat,
    const float* __restrict__ wvec)
{
  __shared__ float Lzx[WW], Lyx[WW];
  const int c = blockIdx.x >> 2, b = blockIdx.x & 3;
  const int w_ = threadIdx.x;
  const float wd = wvec[2];
  const _Float16* src = in + (size_t)c * PT + (size_t)b * PB;
  _Float16* od  = cat + ((size_t)(2 * COUT + c)) * PT + (size_t)b * PB;
  _Float16* ozx = cat + ((size_t)(4 * COUT + c)) * PT + (size_t)b * PB;
  _Float16* oyx = cat + ((size_t)(5 * COUT + c)) * PT + (size_t)b * PB;
  float pd = 0.f;
  for (int h = 0; h < HH; ++h) {
    float cur = (float)src[h * WW + w_];
    float vd, vzx, vyx;
    if (h == 0) { vd = vzx = vyx = cur; }
    else {
      vd  = fmaxf(fmaf(pd, wd, cur), 0.f);
      vzx = (w_ == 0)      ? cur : fmaxf(fmaf(Lzx[w_ - 1], wd, cur), 0.f);
      vyx = (w_ == WW - 1) ? cur : fmaxf(fmaf(Lyx[w_ + 1], wd, cur), 0.f);
    }
    od[h * WW + w_]  = (_Float16)vd;
    ozx[h * WW + w_] = (_Float16)vzx;
    oyx[h * WW + w_] = (_Float16)vyx;
    pd = vd;
    __syncthreads();
    Lzx[w_] = vzx; Lyx[w_] = vyx;
    __syncthreads();
  }
}

// Backward H scan: up (wu). Pure per-thread recurrence, coalesced rows.
__global__ void __launch_bounds__(256) sam_scan_h_bwd(
    const _Float16* __restrict__ in, _Float16* __restrict__ cat,
    const float* __restrict__ wvec)
{
  const int gid = blockIdx.x * 256 + threadIdx.x;   // 32*4*256 lines
  const int c = gid >> 10, rem = gid & 1023;
  const int b = rem >> 8, w_ = rem & 255;
  const float wu = wvec[0];
  const _Float16* src = in + (size_t)c * PT + (size_t)b * PB;
  _Float16* ou = cat + ((size_t)(0 * COUT + c)) * PT + (size_t)b * PB;
  float pv = 0.f;
  for (int hh = 0; hh < HH; ++hh) {
    int h = HH - 1 - hh;
    float cur = (float)src[h * WW + w_];
    float v = (hh == 0) ? cur : fmaxf(fmaf(pv, wu, cur), 0.f);
    ou[h * WW + w_] = (_Float16)v;
    pv = v;
  }
}

// Forward W scan: right (wr), zuoshang (shift +1 over H, wd), youshang (wd).
// 256x64 f16 chunks staged via LDS for coalesced global reads.
__global__ void __launch_bounds__(256) sam_scan_w_fwd(
    const _Float16* __restrict__ in, _Float16* __restrict__ cat,
    const float* __restrict__ wvec)
{
  __shared__ _Float16 tile[HH * 68];   // pitch 68 halves (136B)
  __shared__ float Lzs[HH], Lys[HH];
  const int c = blockIdx.x >> 2, b = blockIdx.x & 3;
  const int t = threadIdx.x;           // owns row h = t during the scan phase
  const float wr = wvec[1], wd = wvec[2];
  const _Float16* src = in + (size_t)c * PT + (size_t)b * PB;
  _Float16* orr = cat + ((size_t)(1 * COUT + c)) * PT + (size_t)b * PB;
  _Float16* ozs = cat + ((size_t)(6 * COUT + c)) * PT + (size_t)b * PB;
  _Float16* oys = cat + ((size_t)(7 * COUT + c)) * PT + (size_t)b * PB;
  float pr_c = 0.f;
  for (int ch = 0; ch < 4; ++ch) {
    const int j0 = ch * 64;
    {
      const int lane = t & 31, r0 = t >> 5;  // wave loads 128B per row
      for (int pass = 0; pass < 32; ++pass) {
        int h = r0 + 8 * pass;
        unsigned int u = *reinterpret_cast<const unsigned int*>(src + h * WW + j0 + lane * 2);
        *reinterpret_cast<unsigned int*>(&tile[h * 68 + lane * 2]) = u;
      }
    }
    __syncthreads();
    for (int jl = 0; jl < 64; ++jl) {
      const int j = j0 + jl;
      float cur = (float)tile[t * 68 + jl];
      float vr, vzs, vys;
      if (j == 0) { vr = vzs = vys = cur; }
      else {
        vr  = fmaxf(fmaf(pr_c, wr, cur), 0.f);
        vzs = (t == 0)      ? cur : fmaxf(fmaf(Lzs[t - 1], wd, cur), 0.f);
        vys = (t == HH - 1) ? cur : fmaxf(fmaf(Lys[t + 1], wd, cur), 0.f);
      }
      orr[t * WW + j] = (_Float16)vr;
      ozs[t * WW + j] = (_Float16)vzs;
      oys[t * WW + j] = (_Float16)vys;
      pr_c = vr;
      __syncthreads();
      Lzs[t] = vzs; Lys[t] = vys;
      __syncthreads();
    }
  }
}

// Backward W scan: left (wl). Chunked LDS staging, per-thread recurrence.
__global__ void __launch_bounds__(256) sam_scan_w_bwd(
    const _Float16* __restrict__ in, _Float16* __restrict__ cat,
    const float* __restrict__ wvec)
{
  __shared__ _Float16 tile[HH * 68];
  const int c = blockIdx.x >> 2, b = blockIdx.x & 3;
  const int t = threadIdx.x;
  const float wl = wvec[3];
  const _Float16* src = in + (size_t)c * PT + (size_t)b * PB;
  _Float16* ol = cat + ((size_t)(3 * COUT + c)) * PT + (size_t)b * PB;
  float pv = 0.f;
  for (int ch = 3; ch >= 0; --ch) {
    const int j0 = ch * 64;
    {
      const int lane = t & 31, r0 = t >> 5;
      for (int pass = 0; pass < 32; ++pass) {
        int h = r0 + 8 * pass;
        unsigned int u = *reinterpret_cast<const unsigned int*>(src + h * WW + j0 + lane * 2);
        *reinterpret_cast<unsigned int*>(&tile[h * 68 + lane * 2]) = u;
      }
    }
    __syncthreads();
    for (int jl = 63; jl >= 0; --jl) {
      const int j = j0 + jl;
      float cur = (float)tile[t * 68 + jl];
      float v = (j == WW - 1) ? cur : fmaxf(fmaf(pv, wl, cur), 0.f);
      ol[t * WW + j] = (_Float16)v;
      pv = v;
    }
    __syncthreads();
  }
}

// ---------------------------------------------------------------------------
extern "C" void kernel_launch(void* const* d_in, const int* in_sizes, int n_in,
                              void* d_out, int out_size, void* d_ws, size_t ws_size,
                              hipStream_t stream) {
  (void)in_sizes; (void)n_in; (void)out_size; (void)ws_size;
  const float* x     = (const float*)d_in[0];
  const float* w1    = (const float*)d_in[1];
  const float* b1    = (const float*)d_in[2];
  const float* w2    = (const float*)d_in[3];
  const float* b2    = (const float*)d_in[4];
  const float* w3    = (const float*)d_in[5];
  const float* b3    = (const float*)d_in[6];
  const float* w_out = (const float*)d_in[7];
  const float* irnn1 = (const float*)d_in[8];
  const float* irnn2 = (const float*)d_in[9];
  float* mask = (float*)d_out;

  // Workspace: cat (8*32 ch, f16, reused for both IRNN stages) + two 32ch f16.
  char* ws = (char*)d_ws;
  _Float16* cat  = (_Float16*)ws;                                  // 128 MB
  _Float16* outA = (_Float16*)(ws + (size_t)8 * COUT * PT * 2);    // 16 MB
  _Float16* outB = outA + (size_t)COUT * PT;                       // 16 MB

  sam_conv1<<<PT / 128, 256, 0, stream>>>(x, w1, b1, outA);

  sam_scan_h_fwd<<<COUT * NB, 256, 0, stream>>>(outA, cat, irnn1);
  sam_scan_h_bwd<<<(COUT * NB * WW) / 256, 256, 0, stream>>>(outA, cat, irnn1);
  sam_scan_w_fwd<<<COUT * NB, 256, 0, stream>>>(outA, cat, irnn1);
  sam_scan_w_bwd<<<COUT * NB, 256, 0, stream>>>(outA, cat, irnn1);

  sam_convk<false><<<PT / 64, 128, 0, stream>>>(cat, w2, b2, nullptr, outB, nullptr);

  sam_scan_h_fwd<<<COUT * NB, 256, 0, stream>>>(outB, cat, irnn2);
  sam_scan_h_bwd<<<(COUT * NB * WW) / 256, 256, 0, stream>>>(outB, cat, irnn2);
  sam_scan_w_fwd<<<COUT * NB, 256, 0, stream>>>(outB, cat, irnn2);
  sam_scan_w_bwd<<<COUT * NB, 256, 0, stream>>>(outB, cat, irnn2);

  sam_convk<true><<<PT / 64, 128, 0, stream>>>(cat, w3, b3, w_out, nullptr, mask);
}
